// EdgeMLPMPN_50414326121238
// MI455X (gfx1250) — matrix operand order repaired and checked by reading, not compiled, not run on hardware
//
#include <hip/hip_runtime.h>

typedef __attribute__((ext_vector_type(16))) _Float16 v16h;
typedef __attribute__((ext_vector_type(8)))  _Float16 v8h;
typedef __attribute__((ext_vector_type(8)))  float    v8f;

#define WAVES     8
#define NTHREADS  (WAVES * 32)

// ---- LDS layout (bytes) ----
// W1 in WMMA-B fragment layout: 8 K-steps * 8 N-tiles * 1KB = 64 KB
// W2 in WMMA-B fragment layout: 4 K-steps * 8 N-tiles * 1KB = 32 KB
// b1 / b2 / w3 as f32: 512 B each
// per-wave A staging: 8 K-step fragments * 1KB = 8 KB per wave
#define W1_OFF   0
#define W2_OFF   (W1_OFF + 64 * 1024)
#define B1_OFF   (W2_OFF + 32 * 1024)
#define B2_OFF   (B1_OFF + 512)
#define W3_OFF   (B2_OFF + 512)
#define A_OFF    (W3_OFF + 512)
#define LDS_BYTES (A_OFF + WAVES * 8192)

__device__ __forceinline__ float elu_f(float v) {
    return v > 0.0f ? v : (__expf(v) - 1.0f);
}

__global__ __launch_bounds__(NTHREADS)
void EdgeMLPMPN_kernel(const float* __restrict__ x,
                       const long long* __restrict__ eidx,
                       const float* __restrict__ W1, const float* __restrict__ b1,
                       const float* __restrict__ W2, const float* __restrict__ b2,
                       const float* __restrict__ W3, const float* __restrict__ b3,
                       float* __restrict__ out, int nEdges)
{
    __shared__ __align__(32) unsigned char smem[LDS_BYTES];
    float* b1s = (float*)(smem + B1_OFF);
    float* b2s = (float*)(smem + B2_OFF);
    float* w3s = (float*)(smem + W3_OFF);

    const int tid  = threadIdx.x;
    const int lane = tid & 31;
    const int wave = tid >> 5;

    // ---- Stage weights (f32 -> f16) into LDS in WMMA-B fragment layout ----
    // B tile (32x16) fragment: lane L holds N = L%16, K = 8*(L/16)+s for s<8, 16+8*(L/16)+(s-8) for s>=8.
    for (int i = tid; i < 128 * 256; i += NTHREADS) {
        int n = i >> 8, k = i & 255;              // W1[n][k], B[k][n] = W1^T
        int s = k >> 5, t = n >> 4, kk = k & 31;
        int slot = (n & 15) + (((kk >> 3) & 1) << 4);
        int h = (kk & 7) + ((kk >> 4) << 3);
        ((_Float16*)(smem + W1_OFF))[(s * 8 + t) * 512 + slot * 16 + h] = (_Float16)W1[i];
    }
    for (int i = tid; i < 128 * 128; i += NTHREADS) {
        int n = i >> 7, k = i & 127;
        int s = k >> 5, t = n >> 4, kk = k & 31;
        int slot = (n & 15) + (((kk >> 3) & 1) << 4);
        int h = (kk & 7) + ((kk >> 4) << 3);
        ((_Float16*)(smem + W2_OFF))[(s * 8 + t) * 512 + slot * 16 + h] = (_Float16)W2[i];
    }
    if (tid < 128) { b1s[tid] = b1[tid]; b2s[tid] = b2[tid]; w3s[tid] = W3[tid]; }
    __syncthreads();

    const float b3v = b3[0];
    const long long* srcI = eidx;
    const long long* tgtI = eidx + nEdges;
    unsigned char* aBuf = smem + A_OFF + wave * 8192;

    const int nTiles = (nEdges + 15) >> 4;
    const int tileStride = gridDim.x * WAVES;

    // precomputed A-layout swizzle pieces for the gather (depend on lane only)
    const int gGrp   = (lane & 15) & 3;                   // K0/8 group within 32-K step
    const int sLane  = ((lane & 15) >> 2) + ((lane >> 4) << 2); // K-step (tgt half -> +4)
    const int h0     = (gGrp >> 1) << 3;                  // starting half index
    const int slotHi = (gGrp & 1) << 4;                   // +16 lane-slot for K in [8,16)|[24,32)

    for (int tile = blockIdx.x * WAVES + wave; tile < nTiles; tile += tileStride) {
        int base = tile << 4;

        // warm L0 for next iteration's edge indices while this tile computes
        int nextTile = tile + tileStride;
        if (nextTile < nTiles) {
            __builtin_prefetch(&srcI[nextTile << 4], 0, 0);
            __builtin_prefetch(&tgtI[nextTile << 4], 0, 0);
        }

        // ---- Gather 16 edges (concat src||tgt features), f16-convert, stage in A layout ----
        // lanes 0-15 fetch src row halves, lanes 16-31 fetch tgt row halves (8 floats/lane/edge)
        for (int m = 0; m < 16; ++m) {
            int e = base + m; if (e >= nEdges) e = nEdges - 1;
            long long row = (lane < 16) ? srcI[e] : tgtI[e];
            const float4* rp = (const float4*)(x + row * 128);
            float4 f0 = rp[(lane & 15) * 2];
            float4 f1 = rp[(lane & 15) * 2 + 1];
            v8h hv;
            hv[0] = (_Float16)f0.x; hv[1] = (_Float16)f0.y;
            hv[2] = (_Float16)f0.z; hv[3] = (_Float16)f0.w;
            hv[4] = (_Float16)f1.x; hv[5] = (_Float16)f1.y;
            hv[6] = (_Float16)f1.z; hv[7] = (_Float16)f1.w;
            *(v8h*)(aBuf + sLane * 1024 + (m + slotHi) * 32 + h0 * 2) = hv;
        }

        // ---- GEMM1: [16 x 256] x [256 x 128] -> c[8] fragments (f32 acc) ----
        // Software-pipelined: issue next A/B fragment loads before the WMMA that
        // consumes the current ones, so the wait is dscnt<=2 instead of dscnt==0.
        v8f c[8];
        #pragma unroll
        for (int t = 0; t < 8; ++t) c[t] = (v8f)0.0f;
        {
            v16h a = *(const v16h*)(aBuf + lane * 32);
            v16h b = *(const v16h*)(smem + W1_OFF + lane * 32);
            #pragma unroll
            for (int s = 0; s < 8; ++s) {
                #pragma unroll
                for (int t = 0; t < 8; ++t) {
                    v16h an = a, bn = b;
                    int nt = t + 1, ns = s;
                    if (nt == 8) { nt = 0; ns = s + 1; }
                    if (ns < 8) {
                        bn = *(const v16h*)(smem + W1_OFF + ((ns * 8 + nt) << 10) + lane * 32);
                        if (nt == 0)
                            an = *(const v16h*)(aBuf + ns * 1024 + lane * 32);
                    }
                    c[t] = __builtin_amdgcn_wmma_f32_16x16x32_f16(
                               false, a, false, b, (short)0, c[t], false, false);
                    a = an; b = bn;
                }
            }
        }

        // ---- bias + ELU, restage h1 (f16) into A layout for layer 2 ----
        #pragma unroll
        for (int t = 0; t < 8; ++t) {
            int n = (lane & 15) + (t << 4);
            float bb = b1s[n];
            int s2 = t >> 1;
            int h  = (lane & 7) + ((t & 1) << 3);
            int sb = (((lane & 15) >> 3) & 1) << 4;
            int Mb = (lane >> 4) << 3;
            #pragma unroll
            for (int r = 0; r < 8; ++r) {
                float v = elu_f(c[t][r] + bb);
                ((_Float16*)(aBuf + s2 * 1024))[(Mb + r + sb) * 16 + h] = (_Float16)v;
            }
        }

        // ---- GEMM2: [16 x 128] x [128 x 128] -> c2[8] fragments (pipelined) ----
        v8f c2[8];
        #pragma unroll
        for (int t = 0; t < 8; ++t) c2[t] = (v8f)0.0f;
        {
            v16h a = *(const v16h*)(aBuf + lane * 32);
            v16h b = *(const v16h*)(smem + W2_OFF + lane * 32);
            #pragma unroll
            for (int s = 0; s < 4; ++s) {
                #pragma unroll
                for (int t = 0; t < 8; ++t) {
                    v16h an = a, bn = b;
                    int nt = t + 1, ns = s;
                    if (nt == 8) { nt = 0; ns = s + 1; }
                    if (ns < 4) {
                        bn = *(const v16h*)(smem + W2_OFF + ((ns * 8 + nt) << 10) + lane * 32);
                        if (nt == 0)
                            an = *(const v16h*)(aBuf + ns * 1024 + lane * 32);
                    }
                    c2[t] = __builtin_amdgcn_wmma_f32_16x16x32_f16(
                                false, a, false, b, (short)0, c2[t], false, false);
                    a = an; b = bn;
                }
            }
        }

        // ---- Layer 3: bias + ELU then dot with w3, reduce across 16-lane halves ----
        float acc[8];
        #pragma unroll
        for (int r = 0; r < 8; ++r) acc[r] = 0.0f;
        #pragma unroll
        for (int t = 0; t < 8; ++t) {
            int n = (lane & 15) + (t << 4);
            float bb = b2s[n];
            float w  = w3s[n];
            #pragma unroll
            for (int r = 0; r < 8; ++r)
                acc[r] += elu_f(c2[t][r] + bb) * w;
        }
        #pragma unroll
        for (int r = 0; r < 8; ++r) {
            acc[r] += __shfl_xor(acc[r], 1);
            acc[r] += __shfl_xor(acc[r], 2);
            acc[r] += __shfl_xor(acc[r], 4);
            acc[r] += __shfl_xor(acc[r], 8);
        }
        if ((lane & 15) == 0) {
            int mbase = base + ((lane >> 4) << 3);
            #pragma unroll
            for (int r = 0; r < 8; ++r)
                if (mbase + r < nEdges) out[mbase + r] = acc[r] + b3v;
        }
    }
}

extern "C" void kernel_launch(void* const* d_in, const int* in_sizes, int n_in,
                              void* d_out, int out_size, void* d_ws, size_t ws_size,
                              hipStream_t stream) {
    const float*     x  = (const float*)d_in[0];
    const long long* ei = (const long long*)d_in[1];
    const float*     W1 = (const float*)d_in[2];
    const float*     b1 = (const float*)d_in[3];
    const float*     W2 = (const float*)d_in[4];
    const float*     b2 = (const float*)d_in[5];
    const float*     W3 = (const float*)d_in[6];
    const float*     b3 = (const float*)d_in[7];
    float* out = (float*)d_out;

    int nEdges = in_sizes[1] / 2;
    int nTiles = (nEdges + 15) / 16;
    // ~4 tiles per wave so the per-block weight staging amortizes
    int blocks = (nTiles + WAVES * 4 - 1) / (WAVES * 4);
    if (blocks < 1) blocks = 1;

    EdgeMLPMPN_kernel<<<blocks, NTHREADS, 0, stream>>>(
        x, ei, W1, b1, W2, b2, W3, b3, out, nEdges);
}
